// WindowAttention_15668040696331
// MI455X (gfx1250) — compile-verified, matrix operand order checked by reading, not run on hardware
//
#include <hip/hip_runtime.h>

typedef __bf16 bf16;
typedef __attribute__((ext_vector_type(16))) __bf16 v16bf;
typedef __attribute__((ext_vector_type(8)))  __bf16 v8bf;
typedef __attribute__((ext_vector_type(8)))  float  v8f;
typedef __attribute__((ext_vector_type(4)))  unsigned int u32x4;
typedef __attribute__((ext_vector_type(8)))  int i32x8;
typedef __attribute__((ext_vector_type(4)))  int i32x4;

#define WIN_N     49
#define NUM_HEADS 4
#define DIM       512
#define HEAD_DIM  128
#define BATCH     2048
#define NUM_WIN   64
#define TOKENS    (BATCH * WIN_N)   // 100352 = 392 * 256

#if defined(__has_builtin)
# if __has_builtin(__builtin_amdgcn_tensor_load_to_lds)
#  define HAVE_TDM 1
# else
#  define HAVE_TDM 0
# endif
#else
# define HAVE_TDM 0
#endif

#if defined(__has_include)
# if __has_include(<hip/amd_detail/amd_gfx1250_TDM.h>)
#  define TDM_6ARG 1            // therock headers -> clang-23 6-arg builtin
# else
#  define TDM_6ARG 0            // ROCm 7.2 clang-22 5-arg builtin
# endif
#else
# define TDM_6ARG 0
#endif

__device__ __forceinline__ bf16 f2bf(float f) {
    unsigned u = __builtin_bit_cast(unsigned, f);
    u += 0x7FFFu + ((u >> 16) & 1u);
    unsigned short h = (unsigned short)(u >> 16);
    return __builtin_bit_cast(bf16, h);
}

__device__ __forceinline__ v16bf ld_pair(const bf16* p0, const bf16* p1) {
    v8bf x = *(const v8bf*)p0;
    v8bf y = *(const v8bf*)p1;
    v16bf r;
#pragma unroll
    for (int i = 0; i < 8; ++i) { r[i] = x[i]; r[i + 8] = y[i]; }
    return r;
}

__device__ __forceinline__ unsigned ldsaddr(const void* p) {
    return (unsigned)(unsigned long long)p;      // LDS offset = addr[31:0]
}

#if HAVE_TDM
// 2D TDM load: rows x cols (bf16 elements, row stride in elements) -> LDS.
// Optional LDS padding: pad_amt_code dwords (code+1) every 2^(pad_int_code+1) dwords.
__device__ __forceinline__ void tdm_load_2d(const void* gptr, unsigned lds_off,
                                            int rows, int cols, int row_stride_elems,
                                            int pad_int_code, int pad_amt_code, int pad_en) {
    unsigned long long ga = (unsigned long long)gptr;
    u32x4 g0;
    g0[0] = 1u;                                                   // count=1, user desc
    g0[1] = lds_off;
    g0[2] = (unsigned)(ga & 0xFFFFFFFFull);
    g0[3] = (unsigned)((ga >> 32) & 0x1FFFFFFull) | (2u << 30);   // type=2
    i32x8 g1;
    g1[0] = (1 << 16) | (pad_en << 20) | (pad_int_code << 22) | (pad_amt_code << 25); // data_size=2B
    g1[1] = (cols & 0xFFFF) << 16;                                // tensor_dim0 lo
    g1[2] = (cols >> 16) | ((rows & 0xFFFF) << 16);               // tensor_dim0 hi | tensor_dim1 lo
    g1[3] = (rows >> 16) | (cols << 16);                          // tensor_dim1 hi | tile_dim0
    g1[4] = rows;                                                 // tile_dim1, tile_dim2=0
    g1[5] = row_stride_elems;                                     // tensor_dim0_stride lo32
    g1[6] = 0;
    g1[7] = 0;
    i32x4 z4 = {0, 0, 0, 0};
#if TDM_6ARG
    i32x8 z8 = {0, 0, 0, 0, 0, 0, 0, 0};
    __builtin_amdgcn_tensor_load_to_lds(g0, g1, z4, z4, z8, 0);
#else
    __builtin_amdgcn_tensor_load_to_lds(g0, g1, z4, z4, 0);
#endif
}
#endif

// ---------------- elementwise fp32 -> bf16 ----------------
__global__ void f32_to_bf16_kernel(const float* __restrict__ in, bf16* __restrict__ out, int n) {
    int i = blockIdx.x * blockDim.x + threadIdx.x;
    if (i < n) out[i] = f2bf(in[i]);
}

// ---------------- fused bias+mask table: bm[h][w][49][49] ----------------
__global__ void bm_precompute_kernel(const float* __restrict__ mask,
                                     const float* __restrict__ bias_table,
                                     const int* __restrict__ rel_index,
                                     float* __restrict__ bm) {
    int idx = blockIdx.x * blockDim.x + threadIdx.x;
    const int NN = WIN_N * WIN_N;                 // 2401
    if (idx >= NUM_HEADS * NUM_WIN * NN) return;
    int h  = idx / (NUM_WIN * NN);
    int w  = (idx / NN) % NUM_WIN;
    int ij = idx % NN;
    bm[idx] = bias_table[rel_index[ij] * NUM_HEADS + h] + mask[w * NN + ij];
}

// ---------------- weight pre-pack (WMMA B-operand per-lane layout) ----------------
__global__ void pack_w_kernel(const float* __restrict__ W, bf16* __restrict__ out, int N, int K) {
    int idx = blockIdx.x * blockDim.x + threadIdx.x;
    if (idx >= N * K) return;
    int ksteps = K >> 5;
    int e    = idx & 15;
    int lane = (idx >> 4) & 31;
    int rem  = idx >> 9;
    int ks   = rem % ksteps;
    int nt   = rem / ksteps;
    int n    = nt * 16 + (lane & 15);
    int kbs  = (lane < 16) ? 0 : 8;
    int kk   = (e < 8) ? (kbs + e) : (16 + kbs + (e - 8));
    out[idx] = f2bf(W[(size_t)n * K + ks * 32 + kk]);
}

// ---------------- GEMM: block = 8 waves = 256x128 supertile; wave = 64x64 ----------------
// Double-buffered TDM staging of A (256x32, 80B padded pitch) and packed B (8x512)
// per K-step: 24KB global traffic per 2.1 MFLOP -> ~87 FLOP/byte.
#define APITCH 40   // LDS A row pitch in elements (32 + 8 pad = 80B)

template <bool OUT_BF16>
__global__ void __launch_bounds__(256)
gemm_wmma(const bf16* __restrict__ A, const bf16* __restrict__ Wpack,
          const float* __restrict__ bias, void* __restrict__ outp,
          int Nblk, int K) {                      // Nblk = N/128
    __shared__ bf16 Abuf[2][256 * APITCH];        // 2 x 20KB
    __shared__ bf16 Bbuf[2][8 * 512];             // 2 x 8KB

    int ksteps = K >> 5;
    int tid  = threadIdx.x;
    int wave = tid >> 5, lane = tid & 31;
    int mt_l = wave >> 1, nt_l = wave & 1;
    int bm_  = blockIdx.x / Nblk, bn = blockIdx.x % Nblk;

    const bf16* Ablk = A + (size_t)bm_ * 256 * K;
    const bf16* Bblk = Wpack + (size_t)bn * 8 * ksteps * 512;

#if HAVE_TDM
    if (tid < 32) {
        tdm_load_2d(Ablk, ldsaddr(Abuf[0]), 256, 32, K, 3, 3, 1);           // +16B pad / 64B row
        tdm_load_2d(Bblk, ldsaddr(Bbuf[0]), 8, 512, ksteps * 512, 0, 0, 0);
    }
#else
    for (int idx = tid; idx < 256 * 32; idx += 256) {
        int r = idx >> 5, c = idx & 31;
        Abuf[0][r * APITCH + c] = Ablk[(size_t)r * K + c];
    }
    for (int idx = tid; idx < 8 * 512; idx += 256) {
        int t = idx >> 9, e = idx & 511;
        Bbuf[0][idx] = Bblk[(size_t)t * ksteps * 512 + e];
    }
#endif

    int kb = (lane < 16) ? 0 : 8;
    v8f acc[4][4] = {};
    for (int ks = 0; ks < ksteps; ++ks) {
        int cur = ks & 1;
#if HAVE_TDM
        if (tid < 32) __builtin_amdgcn_s_wait_tensorcnt(0);
#endif
        __syncthreads();   // publishes buf[cur]; proves buf[cur^1] readers of ks-1 done
        if (ks + 1 < ksteps) {
#if HAVE_TDM
            if (tid < 32) {
                tdm_load_2d(Ablk + (ks + 1) * 32, ldsaddr(Abuf[cur ^ 1]), 256, 32, K, 3, 3, 1);
                tdm_load_2d(Bblk + (size_t)(ks + 1) * 512, ldsaddr(Bbuf[cur ^ 1]), 8, 512,
                            ksteps * 512, 0, 0, 0);
            }
#else
            for (int idx = tid; idx < 256 * 32; idx += 256) {
                int r = idx >> 5, c = idx & 31;
                Abuf[cur ^ 1][r * APITCH + c] = Ablk[(size_t)r * K + (ks + 1) * 32 + c];
            }
            for (int idx = tid; idx < 8 * 512; idx += 256) {
                int t = idx >> 9, e = idx & 511;
                Bbuf[cur ^ 1][idx] = Bblk[((size_t)t * ksteps + ks + 1) * 512 + e];
            }
#endif
        }
        const bf16* Ab = Abuf[cur];
        const bf16* Bb = Bbuf[cur];
        v16bf a[4], b[4];
#pragma unroll
        for (int i = 0; i < 4; ++i) {
            int r = mt_l * 64 + i * 16 + (lane & 15);
            a[i] = ld_pair(Ab + r * APITCH + kb, Ab + r * APITCH + 16 + kb);
        }
#pragma unroll
        for (int j = 0; j < 4; ++j) {
            const bf16* bp = Bb + (nt_l * 4 + j) * 512 + lane * 16;
            b[j] = ld_pair(bp, bp + 8);
        }
#pragma unroll
        for (int i = 0; i < 4; ++i)
#pragma unroll
            for (int j = 0; j < 4; ++j)
                acc[i][j] = __builtin_amdgcn_wmma_f32_16x16x32_bf16(
                    false, a[i], false, b[j], (short)0, acc[i][j], false, false);
    }

    int N = Nblk * 128;
#pragma unroll
    for (int j = 0; j < 4; ++j) {
        int col = bn * 128 + nt_l * 64 + j * 16 + (lane & 15);
        float bv = bias[col];
#pragma unroll
        for (int i = 0; i < 4; ++i) {
            int rbase = bm_ * 256 + mt_l * 64 + i * 16 + ((lane < 16) ? 0 : 8);
#pragma unroll
            for (int r = 0; r < 8; ++r) {
                float v = acc[i][j][r] + bv;
                size_t o = (size_t)(rbase + r) * N + col;
                if (OUT_BF16) ((bf16*)outp)[o] = f2bf(v);
                else          ((float*)outp)[o] = v;
            }
        }
    }
}

// ---------------- attention core: one workgroup per (window, head) ----------------
__global__ void __launch_bounds__(128)
attn_wmma(const bf16* __restrict__ qkv, const float* __restrict__ bmt,
          bf16* __restrict__ attn_out) {
    __shared__ bf16  Qs[64 * 128];
    __shared__ bf16  Ks[64 * 128];
    __shared__ bf16  Vs[64 * 128];
    __shared__ bf16  Vt[128 * 64];
    __shared__ float Sf[64 * 64];
    __shared__ bf16  Ps[64 * 64];

    int b = blockIdx.x >> 2;
    int h = blockIdx.x & 3;
    int w = b & (NUM_WIN - 1);
    int tid = threadIdx.x;

    const size_t rs = 3 * DIM;
    const bf16* qb = qkv + (size_t)b * WIN_N * rs + h * HEAD_DIM;
    const bf16* kp = qb + DIM;
    const bf16* vp = qb + 2 * DIM;
    const float* bmhw = bmt + ((size_t)h * NUM_WIN + w) * WIN_N * WIN_N;

#if HAVE_TDM
    if (tid < 32) {
        tdm_load_2d(qb, ldsaddr(Qs), WIN_N, HEAD_DIM, 3 * DIM, 0, 0, 0);
        tdm_load_2d(kp, ldsaddr(Ks), WIN_N, HEAD_DIM, 3 * DIM, 0, 0, 0);
        tdm_load_2d(vp, ldsaddr(Vs), WIN_N, HEAD_DIM, 3 * DIM, 0, 0, 0);
        __builtin_amdgcn_s_wait_tensorcnt(0);
    }
    for (int idx = tid; idx < 15 * 128; idx += 128) {  // zero pad rows 49..63
        int m = 49 + (idx >> 7), d = idx & 127;
        bf16 z = f2bf(0.f);
        Qs[m * 128 + d] = z; Ks[m * 128 + d] = z; Vs[m * 128 + d] = z;
    }
#else
    for (int idx = tid; idx < 64 * 128; idx += 128) {
        int m = idx >> 7, d = idx & 127;
        bf16 qv, kv, vv;
        if (m < WIN_N) {
            qv = qb[(size_t)m * rs + d];
            kv = kp[(size_t)m * rs + d];
            vv = vp[(size_t)m * rs + d];
        } else { qv = f2bf(0.f); kv = qv; vv = qv; }
        Qs[idx] = qv; Ks[idx] = kv; Vs[idx] = vv;
    }
#endif
    __syncthreads();

    for (int idx = tid; idx < 64 * 128; idx += 128) {  // V transpose for P*V B-operand
        int m = idx >> 7, d = idx & 127;
        Vt[d * 64 + m] = Vs[idx];
    }
    __syncthreads();

    int wave = tid >> 5, lane = tid & 31;
    int kb   = (lane < 16) ? 0 : 8;
    int M0   = wave * 16;
    int arow = M0 + (lane & 15);
    const float scale = 0.08838834764831845f;

    for (int ct = 0; ct < 4; ++ct) {                    // S = Q*K^T
        int bcol = ct * 16 + (lane & 15);
        v8f acc = {};
#pragma unroll
        for (int ks = 0; ks < 4; ++ks) {
            int k0 = ks * 32;
            v16bf a  = ld_pair(&Qs[arow * 128 + k0 + kb], &Qs[arow * 128 + k0 + 16 + kb]);
            v16bf bm = ld_pair(&Ks[bcol * 128 + k0 + kb], &Ks[bcol * 128 + k0 + 16 + kb]);
            acc = __builtin_amdgcn_wmma_f32_16x16x32_bf16(false, a, false, bm, (short)0, acc, false, false);
        }
        int rbase = M0 + ((lane < 16) ? 0 : 8);
#pragma unroll
        for (int r = 0; r < 8; ++r) {
            int rr = rbase + r;
            float v = (rr < WIN_N && bcol < WIN_N)
                    ? acc[r] * scale + bmhw[rr * WIN_N + bcol]
                    : -1e30f;
            Sf[rr * 64 + bcol] = v;
        }
    }
    __syncthreads();

    if (tid < 64) {                                     // row softmax
        if (tid < WIN_N) {
            float mx = -1e30f;
            for (int j = 0; j < WIN_N; ++j) mx = fmaxf(mx, Sf[tid * 64 + j]);
            float s = 0.f;
            for (int j = 0; j < WIN_N; ++j) {
                float e = __expf(Sf[tid * 64 + j] - mx);
                s += e; Sf[tid * 64 + j] = e;
            }
            float inv = 1.f / s;
            for (int j = 0; j < 64; ++j)
                Ps[tid * 64 + j] = (j < WIN_N) ? f2bf(Sf[tid * 64 + j] * inv) : f2bf(0.f);
        } else {
            for (int j = 0; j < 64; ++j) Ps[tid * 64 + j] = f2bf(0.f);
        }
    }
    __syncthreads();

    for (int ct = 0; ct < 8; ++ct) {                    // O = P*V
        int dcol = ct * 16 + (lane & 15);
        v8f acc = {};
#pragma unroll
        for (int ks = 0; ks < 2; ++ks) {
            int k0 = ks * 32;
            v16bf a  = ld_pair(&Ps[arow * 64 + k0 + kb], &Ps[arow * 64 + k0 + 16 + kb]);
            v16bf bm = ld_pair(&Vt[dcol * 64 + k0 + kb], &Vt[dcol * 64 + k0 + 16 + kb]);
            acc = __builtin_amdgcn_wmma_f32_16x16x32_bf16(false, a, false, bm, (short)0, acc, false, false);
        }
        int rbase = M0 + ((lane < 16) ? 0 : 8);
#pragma unroll
        for (int r = 0; r < 8; ++r) {
            int rr = rbase + r;
            if (rr < WIN_N)
                attn_out[((size_t)b * WIN_N + rr) * DIM + h * HEAD_DIM + dcol] = f2bf(acc[r]);
        }
    }
}

extern "C" void kernel_launch(void* const* d_in, const int* in_sizes, int n_in,
                              void* d_out, int out_size, void* d_ws, size_t ws_size,
                              hipStream_t stream) {
    const float* x          = (const float*)d_in[0];
    const float* mask       = (const float*)d_in[1];
    const float* qkv_w      = (const float*)d_in[2];
    const float* qkv_b      = (const float*)d_in[3];
    const float* proj_w     = (const float*)d_in[4];
    const float* proj_b     = (const float*)d_in[5];
    const float* bias_table = (const float*)d_in[6];
    const int*   rel_index  = (const int*)d_in[7];

    char* ws = (char*)d_ws;
    size_t off = 0;
    bf16*  Xb  = (bf16*)(ws + off);  off += (size_t)TOKENS * DIM * 2;       // 103 MB
    bf16*  Wq  = (bf16*)(ws + off);  off += (size_t)3 * DIM * DIM * 2;      // 1.5 MB
    bf16*  Wp  = (bf16*)(ws + off);  off += (size_t)DIM * DIM * 2;          // 0.5 MB
    bf16*  QKV = (bf16*)(ws + off);  off += (size_t)TOKENS * 3 * DIM * 2;   // 308 MB
    bf16*  AO  = (bf16*)(ws + off);  off += (size_t)TOKENS * DIM * 2;       // 103 MB
    float* BMT = (float*)(ws + off); off += (size_t)NUM_HEADS * NUM_WIN * WIN_N * WIN_N * 4; // 2.5 MB

    {   int n = TOKENS * DIM;
        f32_to_bf16_kernel<<<(n + 255) / 256, 256, 0, stream>>>(x, Xb, n); }
    {   int n = 3 * DIM * DIM;
        pack_w_kernel<<<(n + 255) / 256, 256, 0, stream>>>(qkv_w, Wq, 3 * DIM, DIM);
        int n2 = DIM * DIM;
        pack_w_kernel<<<(n2 + 255) / 256, 256, 0, stream>>>(proj_w, Wp, DIM, DIM); }
    {   int n = NUM_HEADS * NUM_WIN * WIN_N * WIN_N;
        bm_precompute_kernel<<<(n + 255) / 256, 256, 0, stream>>>(mask, bias_table, rel_index, BMT); }

    {   // QKV projection -> bf16 [100352,1536]
        int Nblk = (3 * DIM) / 128;                  // 12
        gemm_wmma<true><<<(TOKENS / 256) * Nblk, 256, 0, stream>>>(Xb, Wq, qkv_b, (void*)QKV, Nblk, DIM); }

    attn_wmma<<<BATCH * NUM_HEADS, 128, 0, stream>>>(QKV, BMT, AO);

    {   // output projection -> fp32 d_out [100352,512]
        int Nblk = DIM / 128;                        // 4
        gemm_wmma<false><<<(TOKENS / 256) * Nblk, 256, 0, stream>>>(AO, Wp, proj_b, d_out, Nblk, DIM); }
}
